// EarthSpecificBlock_89558658056661
// MI455X (gfx1250) — matrix operand order repaired and checked by reading, compiled, not run on hardware
//
#include <hip/hip_runtime.h>
#include <math.h>

// ---------------------------------------------------------------------------
// EarthSpecificBlock (Pangu-Weather 3D shifted-window attention block)
// MI455X / gfx1250, wave32.  v_wmma_f32_16x16x32_bf16 for all GEMMs,
// global_load_async_to_lds_b128 double-buffered tile staging (ASYNCcnt).
// ---------------------------------------------------------------------------

typedef __attribute__((ext_vector_type(16))) __bf16 v16bf;
typedef __attribute__((ext_vector_type(8)))  __bf16 v8bf;
typedef __attribute__((ext_vector_type(8)))  float  v8f;

#define DIMC   192
#define HEADS  6
#define HD     32
#define WZ     2
#define WH     6
#define WW     12
#define NWIN   144           // WZ*WH*WW
#define ZG     8
#define HG     96
#define WG     180
#define NWZ    4             // ZG/WZ
#define NWH    16            // HG/WH
#define NWW    15            // WG/WW
#define NWINDOWS 960         // NWZ*NWH*NWW
#define MTOK   138240        // ZG*HG*WG
#define QSCALE 0.17677669529663687f  // 1/sqrt(32)

static __device__ inline v8f wmma_bf16(v16bf a, v16bf b, v8f c) {
  return __builtin_amdgcn_wmma_f32_16x16x32_bf16(false, a, false, b,
                                                 (short)0, c, false, false);
}

union FragBF { v16bf v; v8bf h[2]; };

// A-fragment (16x32 bf16): lane l holds row m=l&15; lanes<16 -> K{0..7,16..23},
// lanes>=16 -> K{8..15,24..31}. Two 16B reads from row-major [m][ld] storage.
static __device__ inline v16bf frag_a16(const __bf16* base, int row0, int ld, int kofs) {
  const int lane = threadIdx.x & 31;
  const int m  = row0 + (lane & 15);
  const int ks = kofs + ((lane & 16) ? 8 : 0);
  FragBF f;
  f.h[0] = *(const v8bf*)(base + (size_t)m * ld + ks);
  f.h[1] = *(const v8bf*)(base + (size_t)m * ld + ks + 16);
  return f.v;
}

// B-fragment (32x16 bf16): lane l holds col n=l&15; lanes<16 -> K{0..7,16..23},
// lanes>=16 -> K{8..15,24..31}.  Two 16B reads from n-major [n][ld] storage.
static __device__ inline v16bf frag_b16(const __bf16* baseT, int n0, int ld, int kofs) {
  const int lane = threadIdx.x & 31;
  const int n  = n0 + (lane & 15);
  const int ks = kofs + ((lane & 16) ? 8 : 0);
  FragBF f;
  f.h[0] = *(const v8bf*)(baseT + (size_t)n * ld + ks);
  f.h[1] = *(const v8bf*)(baseT + (size_t)n * ld + ks + 16);
  return f.v;
}

// Async 16B global -> LDS copy (CDNA5, tracked by ASYNCcnt).  Flat LDS
// aperture keeps the LDS byte address in addr[31:0] (ISA 10.2), so the
// truncated generic pointer is the VDST LDS address.
static __device__ inline void async_copy16(void* lds, const void* g) {
  unsigned l = (unsigned)(size_t)lds;
  asm volatile("global_load_async_to_lds_b128 %0, %1, off"
               :: "v"(l), "v"(g) : "memory");
}
static __device__ inline void async_wait0() {
  asm volatile("s_wait_asynccnt 0x0" ::: "memory");
}

// ---------------------------------------------------------------------------
// fp32 (K,N) -> bf16 (N,K) weight convert+transpose (n-major B for the GEMMs)
// ---------------------------------------------------------------------------
__global__ void cvt_w_bf16t(const float* __restrict__ s, __bf16* __restrict__ d,
                            int K, int N) {
  int i = blockIdx.x * 256 + threadIdx.x;
  if (i < K * N) {
    int k = i / N, n = i - k * N;
    d[(size_t)n * K + k] = (__bf16)s[i];
  }
}

// ---------------------------------------------------------------------------
// LayerNorm (C=192).  gather!=0: fuse roll(-1,-3,-6) + window partition,
// writing bf16 rows in window order for the QKV GEMM.
// ---------------------------------------------------------------------------
__global__ __launch_bounds__(64)
void ln_kernel(const float* __restrict__ xin, const float* __restrict__ w,
               const float* __restrict__ b, __bf16* __restrict__ out,
               const int* __restrict__ rollp, int gather) {
  __shared__ float r1[64], r2[64];
  const int t = blockIdx.x, tid = threadIdx.x;
  float v[3];
#pragma unroll
  for (int j = 0; j < 3; ++j) v[j] = xin[(size_t)t * DIMC + tid + j * 64];
  r1[tid] = v[0] + v[1] + v[2];
  r2[tid] = v[0]*v[0] + v[1]*v[1] + v[2]*v[2];
  __syncthreads();
  for (int st = 32; st > 0; st >>= 1) {
    if (tid < st) { r1[tid] += r1[tid + st]; r2[tid] += r2[tid + st]; }
    __syncthreads();
  }
  const float mean = r1[0] * (1.0f / DIMC);
  const float var  = r2[0] * (1.0f / DIMC) - mean * mean;
  const float rstd = rsqrtf(var + 1e-5f);

  size_t row = (size_t)t;
  if (gather) {
    const int roll = *rollp;
    int zo = t / (HG * WG), ho = (t / WG) % HG, wo = t % WG;
    int zr = roll ? (zo + ZG - 1) % ZG : zo;   // rolled coord i: orig = (i+s)%D
    int hr = roll ? (ho + HG - 3) % HG : ho;
    int wr = roll ? (wo + WG - 6) % WG : wo;
    int bz = zr / WZ, z = zr % WZ;
    int bh = hr / WH, h = hr % WH;
    int bw = wr / WW, ww = wr % WW;
    int win = (bz * NWH + bh) * NWW + bw;
    int nn  = (z * WH + h) * WW + ww;
    row = (size_t)win * NWIN + nn;
  }
#pragma unroll
  for (int j = 0; j < 3; ++j) {
    int c = tid + j * 64;
    float y = (v[j] - mean) * rstd * w[c] + b[c];
    out[row * DIMC + c] = (__bf16)y;
  }
}

// ---------------------------------------------------------------------------
// Generic bf16 WMMA GEMM: C(M,N) = A(M,K) @ Bt(N,K)^T + bias, fused epilogues.
// Block tile 128x64, K-step 32, 4 waves (each 32x64 -> 8 wmma / K-step).
// Double-buffered LDS tiles filled with global_load_async_to_lds_b128 so the
// copy of tile i+1 overlaps the WMMAs of tile i.
// EPI 0: QKV (split q/k/v, scale q) | 1: proj (+reverse roll scatter+residual)
// EPI 2: fc1 (+GELU, bf16 out)      | 3: fc2 (+residual, f32 out)
// ---------------------------------------------------------------------------
template <int EPI>
__global__ __launch_bounds__(128)
void gemm_bf16_kernel(const __bf16* __restrict__ A, const __bf16* __restrict__ Bt,
                      const float* __restrict__ bias, int M, int N, int K,
                      float* __restrict__ outF, __bf16* __restrict__ outH,
                      const float* __restrict__ addF, const int* __restrict__ rollp) {
  __shared__ __bf16 sA[2][128 * 32];   // row-major [m][k]
  __shared__ __bf16 sBt[2][64 * 32];   // n-major   [n][k]
  const int tid  = threadIdx.x;
  const int lane = tid & 31;
  const int wave = tid >> 5;
  const int bm = blockIdx.x * 128;
  const int bn = blockIdx.y * 64;

  v8f acc[2][4];
  const v8f vz = {0.f,0.f,0.f,0.f,0.f,0.f,0.f,0.f};
#pragma unroll
  for (int a = 0; a < 2; ++a)
#pragma unroll
    for (int c = 0; c < 4; ++c) acc[a][c] = vz;

  const int T = K >> 5;  // K/32 tiles

  // issue async copy of K-tile `kt` into buffer `buf`
  auto issue_tile = [&](int kt, int buf) {
    const int k0 = kt * 32;
#pragma unroll
    for (int c = tid; c < 512; c += 128) {            // A: 128 rows x 4 chunks
      int row = c >> 2, ch = c & 3;
      async_copy16(&sA[buf][row * 32 + ch * 8],
                   A + (size_t)(bm + row) * K + k0 + ch * 8);
    }
#pragma unroll
    for (int c = tid; c < 256; c += 128) {            // Bt: 64 rows x 4 chunks
      int nr = c >> 2, ch = c & 3;
      async_copy16(&sBt[buf][nr * 32 + ch * 8],
                   Bt + (size_t)(bn + nr) * K + k0 + ch * 8);
    }
  };

  issue_tile(0, 0);
  async_wait0();
  __syncthreads();

  for (int i = 0; i < T; ++i) {
    const int cur = i & 1;
    if (i + 1 < T) issue_tile(i + 1, 1 - cur);        // overlap copy w/ wmma

    v16bf af[2], bfg[4];
    af[0] = frag_a16(sA[cur], wave * 32,      32, 0);
    af[1] = frag_a16(sA[cur], wave * 32 + 16, 32, 0);
#pragma unroll
    for (int ns = 0; ns < 4; ++ns) bfg[ns] = frag_b16(sBt[cur], ns * 16, 32, 0);
#pragma unroll
    for (int ms = 0; ms < 2; ++ms)
#pragma unroll
      for (int ns = 0; ns < 4; ++ns)
        acc[ms][ns] = wmma_bf16(af[ms], bfg[ns], acc[ms][ns]);

    if (i + 1 < T) async_wait0();                     // tile i+1 landed
    __syncthreads();
  }

  const int roll = rollp ? *rollp : 0;
#pragma unroll
  for (int ms = 0; ms < 2; ++ms)
#pragma unroll
    for (int ns = 0; ns < 4; ++ns)
#pragma unroll
      for (int i = 0; i < 8; ++i) {
        int m = bm + wave * 32 + ms * 16 + i + ((lane & 16) ? 8 : 0);
        int n = bn + ns * 16 + (lane & 15);
        float val = acc[ms][ns][i] + bias[n];
        if (EPI == 0) {                    // QKV: split + scale q
          int sel = n / DIMC, cc = n - sel * DIMC;
          float s = (sel == 0) ? QSCALE : 1.0f;
          outH[(size_t)sel * MTOK * DIMC + (size_t)m * DIMC + cc] = (__bf16)(val * s);
        } else if (EPI == 1) {             // proj: reverse roll + shortcut
          int win = m / NWIN, nn = m % NWIN;
          int bw2 = win % NWW, bh2 = (win / NWW) % NWH, bz2 = win / (NWW * NWH);
          int z = nn / (WH * WW), h = (nn / WW) % WH, w2 = nn % WW;
          int zr = bz2 * WZ + z, hr = bh2 * WH + h, wr = bw2 * WW + w2;
          int zo = roll ? (zr + 1) % ZG : zr;
          int ho = roll ? (hr + 3) % HG : hr;
          int wo = roll ? (wr + 6) % WG : wr;
          size_t tok = ((size_t)zo * HG + ho) * WG + wo;
          outF[tok * DIMC + n] = addF[tok * DIMC + n] + val;
        } else if (EPI == 2) {             // fc1: exact-erf GELU
          float g = 0.5f * val * (1.0f + erff(val * 0.70710678118654752f));
          outH[(size_t)m * 768 + n] = (__bf16)g;
        } else {                           // fc2: residual, f32 out
          outF[(size_t)m * DIMC + n] = addF[(size_t)m * DIMC + n] + val;
        }
      }
}

// ---------------------------------------------------------------------------
// Attention: one workgroup per (window, head), 9 waves (288 threads).
// S = qk^T (81 wmma), in-kernel rel-pos bias + shift mask, register softmax
// (16-lane shfl_xor reductions), P staged to LDS (aliasing dead Q/K space),
// O = P@V with K padded 144->160 (10 wmma/wave).  Static LDS: 56320 B.
// Q/K tiles staged with async-LDS copies.
// ---------------------------------------------------------------------------
__global__ __launch_bounds__(288)
void attn_kernel(const __bf16* __restrict__ qkv, const float* __restrict__ bias_table,
                 __bf16* __restrict__ outO, const int* __restrict__ rollp) {
  __shared__ __align__(16) char smem[56320];
  __bf16* sQ  = (__bf16*)smem;                 // 144x32  (phase 1)
  __bf16* sK  = (__bf16*)(smem + 9216);        // 144x32  (phase 1)
  __bf16* sP  = (__bf16*)smem;                 // 144x160 (phase 2, aliases Q/K)
  __bf16* sVt = (__bf16*)(smem + 46080);       // 32x160  n-major, K zero-padded

  const int tid = threadIdx.x, lane = tid & 31, wave = tid >> 5;
  const int win = blockIdx.x, head = blockIdx.y;
  const size_t QS = (size_t)MTOK * DIMC;
  const __bf16* qb = qkv + (size_t)win * NWIN * DIMC + head * HD;
  const __bf16* kb = qb + QS;
  const __bf16* vb = qb + 2 * QS;

  for (int c = tid; c < 576; c += 288) {        // 144 rows x 4 chunks of 16B
    int r = c >> 2, ch = c & 3;
    async_copy16(&sQ[r * 32 + ch * 8], qb + (size_t)r * DIMC + ch * 8);
    async_copy16(&sK[r * 32 + ch * 8], kb + (size_t)r * DIMC + ch * 8);
  }
  for (int e = tid; e < NWIN * HD; e += 288) {  // V transposed to n-major
    int r = e >> 5, hd = e & 31;
    sVt[hd * 160 + r] = vb[(size_t)r * DIMC + hd];
  }
  for (int e = tid; e < HD * 16; e += 288) {    // zero-pad K rows 144..159
    int hd = e >> 4, r = 144 + (e & 15);
    sVt[hd * 160 + r] = (__bf16)0.0f;
  }
  async_wait0();
  __syncthreads();

  // ---- S = q @ k^T  (K row-major LDS == B^T layout for k^T) ----
  const int tm = wave;                          // M-tile 0..8
  v8f s[9];
  const v8f vz = {0.f,0.f,0.f,0.f,0.f,0.f,0.f,0.f};
  {
    v16bf aq = frag_a16(sQ, tm * 16, 32, 0);
#pragma unroll
    for (int tn = 0; tn < 9; ++tn)
      s[tn] = wmma_bf16(aq, frag_b16(sK, tn * 16, 32, 0), vz);
  }

  // ---- relative-position bias + shift mask ----
  const int roll = *rollp;
  const int bz = win / (NWW * NWH), bh = (win / NWW) % NWH, bw = win % NWW;
#pragma unroll
  for (int tn = 0; tn < 9; ++tn) {
    int n = tn * 16 + (lane & 15);
    int zn = n / (WH * WW), hn = (n / WW) % WH, wn = n % WW;
    int cntn = 0;
    if (roll) {   // region ids reproduce the reference's wz/sz slice quirk
      int zg = bz * WZ + zn, hg = bh * WH + hn, wg = bw * WW + wn;
      int rz = (zg >= ZG - 1) ? 2 : ((zg >= ZG - 2) ? 1 : 0);
      int rh = (hg >= HG - 1) ? 2 : ((hg >= HG - 2) ? 1 : 0);
      int rw = (wg >= WG - 1) ? 2 : ((wg >= WG - 2) ? 1 : 0);
      cntn = rz * 9 + rh * 3 + rw;
    }
#pragma unroll
    for (int i = 0; i < 8; ++i) {
      int m = tm * 16 + i + ((lane & 16) ? 8 : 0);
      int zm = m / (WH * WW), hm = (m / WW) % WH, wm = m % WW;
      int idx = (zm - zn + WZ - 1) * ((2*WH-1)*(2*WW-1))
              + (hm - hn + WH - 1) * (2*WW-1)
              + (wm - wn + WW - 1);
      float add = bias_table[idx * HEADS + head];
      if (roll) {
        int zg = bz * WZ + zm, hg = bh * WH + hm, wg = bw * WW + wm;
        int rz = (zg >= ZG - 1) ? 2 : ((zg >= ZG - 2) ? 1 : 0);
        int rh = (hg >= HG - 1) ? 2 : ((hg >= HG - 2) ? 1 : 0);
        int rw = (wg >= WG - 1) ? 2 : ((wg >= WG - 2) ? 1 : 0);
        if (rz * 9 + rh * 3 + rw != cntn) add += -100.0f;
      }
      s[tn][i] += add;
    }
  }

  __syncthreads();   // all waves done reading sQ/sK before sP aliases them

  // ---- softmax per row (values for row i live in 16 lanes of one half) ----
#pragma unroll
  for (int i = 0; i < 8; ++i) {
    float mx = -3.0e38f;
#pragma unroll
    for (int tn = 0; tn < 9; ++tn) mx = fmaxf(mx, s[tn][i]);
#pragma unroll
    for (int o = 1; o < 16; o <<= 1) mx = fmaxf(mx, __shfl_xor(mx, o, 32));
    float sum = 0.0f;
#pragma unroll
    for (int tn = 0; tn < 9; ++tn) {
      float e = __expf(s[tn][i] - mx);
      s[tn][i] = e; sum += e;
    }
#pragma unroll
    for (int o = 1; o < 16; o <<= 1) sum += __shfl_xor(sum, o, 32);
    float inv = 1.0f / sum;
    int m = tm * 16 + i + ((lane & 16) ? 8 : 0);
#pragma unroll
    for (int tn = 0; tn < 9; ++tn)
      sP[m * 160 + tn * 16 + (lane & 15)] = (__bf16)(s[tn][i] * inv);
  }
  for (int e = tid; e < NWIN * 16; e += 288) {  // zero-pad P cols 144..159
    int r = e >> 4, cp = 144 + (e & 15);
    sP[r * 160 + cp] = (__bf16)0.0f;
  }
  __syncthreads();

  // ---- O = P @ V  (K = 160, 5 steps of 32) ----
  v8f o0 = vz, o1 = vz;
#pragma unroll
  for (int kk = 0; kk < 5; ++kk) {
    v16bf ap = frag_a16(sP, tm * 16, 160, kk * 32);
    o0 = wmma_bf16(ap, frag_b16(sVt, 0,  160, kk * 32), o0);
    o1 = wmma_bf16(ap, frag_b16(sVt, 16, 160, kk * 32), o1);
  }
#pragma unroll
  for (int i = 0; i < 8; ++i) {
    int m = tm * 16 + i + ((lane & 16) ? 8 : 0);
    size_t rowg = (size_t)win * NWIN + m;
    outO[rowg * DIMC + head * HD + (lane & 15)]      = (__bf16)o0[i];
    outO[rowg * DIMC + head * HD + 16 + (lane & 15)] = (__bf16)o1[i];
  }
}

// ---------------------------------------------------------------------------
// Host-side launch sequence (graph-capture safe: launches only).
// Workspace layout (bytes):
//   [0,53M)      win bf16  -> reused as attention output O bf16
//   [53M,212M)   q/k/v bf16 (3x53M) -> region reused: x2 f32 at 53M after attn
//   [159M,212M)  xn bf16 (LN2 out)
//   [212M,425M)  h bf16 (fc1 out, 138240x768)
//   [425M,...)   bf16 transposed weights
// ---------------------------------------------------------------------------
extern "C" void kernel_launch(void* const* d_in, const int* in_sizes, int n_in,
                              void* d_out, int out_size, void* d_ws, size_t ws_size,
                              hipStream_t stream) {
  (void)in_sizes; (void)n_in; (void)out_size; (void)ws_size;
  const float* x        = (const float*)d_in[0];
  const float* qkv_w    = (const float*)d_in[1];
  const float* qkv_b    = (const float*)d_in[2];
  const float* proj_w   = (const float*)d_in[3];
  const float* proj_b   = (const float*)d_in[4];
  const float* bias_tab = (const float*)d_in[5];
  const float* n1w      = (const float*)d_in[6];
  const float* n1b      = (const float*)d_in[7];
  const float* n2w      = (const float*)d_in[8];
  const float* n2b      = (const float*)d_in[9];
  const float* fc1_w    = (const float*)d_in[10];
  const float* fc1_b    = (const float*)d_in[11];
  const float* fc2_w    = (const float*)d_in[12];
  const float* fc2_b    = (const float*)d_in[13];
  const int*   rollp    = (const int*)d_in[17];

  char* ws = (char*)d_ws;
  const size_t SZ_TOKC_BF = (size_t)MTOK * DIMC * 2;      //  53,084,160
  __bf16* wWin  = (__bf16*)(ws + 0);
  __bf16* wO    = (__bf16*)(ws + 0);                      // reuse after QKV
  __bf16* wQKV  = (__bf16*)(ws + SZ_TOKC_BF);             // 3 x tokc bf16
  float*  wX2   = (float*) (ws + SZ_TOKC_BF);             // reuse after attn
  __bf16* wXn   = (__bf16*)(ws + 3 * SZ_TOKC_BF);
  __bf16* wH    = (__bf16*)(ws + 4 * SZ_TOKC_BF);         // 138240x768 bf16
  char*   wWts  = ws + 8 * SZ_TOKC_BF;
  __bf16* wQW  = (__bf16*)(wWts);                          // (576,192)
  __bf16* wPW  = (__bf16*)(wWts + 221184);                 // (192,192)
  __bf16* wF1W = (__bf16*)(wWts + 221184 + 73728);         // (768,192)
  __bf16* wF2W = (__bf16*)(wWts + 221184 + 73728 + 294912);// (192,768)

  // 1) weights -> bf16, transposed to n-major (N,K) for async B-tile staging
  cvt_w_bf16t<<<(110592 + 255) / 256, 256, 0, stream>>>(qkv_w, wQW, DIMC, 576);
  cvt_w_bf16t<<<(36864  + 255) / 256, 256, 0, stream>>>(proj_w, wPW, DIMC, DIMC);
  cvt_w_bf16t<<<(147456 + 255) / 256, 256, 0, stream>>>(fc1_w, wF1W, DIMC, 768);
  cvt_w_bf16t<<<(147456 + 255) / 256, 256, 0, stream>>>(fc2_w, wF2W, 768, DIMC);

  // 2) LN1 + roll + window partition (bf16)
  ln_kernel<<<MTOK, 64, 0, stream>>>(x, n1w, n1b, wWin, rollp, 1);

  // 3) QKV GEMM: (138240,192)@(192,576)
  gemm_bf16_kernel<0><<<dim3(MTOK / 128, 576 / 64), 128, 0, stream>>>(
      wWin, wQW, qkv_b, MTOK, 576, DIMC, nullptr, wQKV, nullptr, rollp);

  // 4) windowed attention (960 windows x 6 heads)
  attn_kernel<<<dim3(NWINDOWS, HEADS), 288, 0, stream>>>(wQKV, bias_tab, wO, rollp);

  // 5) proj GEMM + reverse roll scatter + shortcut residual -> x2 (f32)
  gemm_bf16_kernel<1><<<dim3(MTOK / 128, DIMC / 64), 128, 0, stream>>>(
      wO, wPW, proj_b, MTOK, DIMC, DIMC, wX2, nullptr, x, rollp);

  // 6) LN2 -> bf16
  ln_kernel<<<MTOK, 64, 0, stream>>>(wX2, n2w, n2b, wXn, rollp, 0);

  // 7) fc1 GEMM + GELU: (138240,192)@(192,768)
  gemm_bf16_kernel<2><<<dim3(MTOK / 128, 768 / 64), 128, 0, stream>>>(
      wXn, wF1W, fc1_b, MTOK, 768, DIMC, nullptr, wH, nullptr, rollp);

  // 8) fc2 GEMM + residual: (138240,768)@(768,192) -> d_out (f32)
  gemm_bf16_kernel<3><<<dim3(MTOK / 128, DIMC / 64), 128, 0, stream>>>(
      wH, wF2W, fc2_b, MTOK, DIMC, 768, (float*)d_out, nullptr, wX2, rollp);
}
